// CrossAssetAttentionNetwork_84112639525707
// MI455X (gfx1250) — matrix operand order repaired and checked by reading, not compiled, not run on hardware
//
#include <hip/hip_runtime.h>
#include <hip/hip_bf16.h>
#include <math.h>

// ---------------------------------------------------------------------------
// CrossAssetAttentionNetwork for MI455X (gfx1250, wave32, WMMA bf16)
//   k0: convert Wq/Wk/Wv fp32 -> bf16
//   k1: fused QKV projection, one A-fragment feeds 12 WMMAs per K-step,
//       V written transposed [B, DOUT, N] via LDS staging (b128 stores)
//   k2: flash attention: rank-gate LUT, online softmax, double-buffered
//       fragments, global_prefetch of the next key block, Ws head fused
// ---------------------------------------------------------------------------

typedef __bf16 bf16;
typedef __attribute__((ext_vector_type(16))) __bf16 v16bf;
typedef __attribute__((ext_vector_type(8)))  __bf16 v8bf;
typedef __attribute__((ext_vector_type(8)))  float  v8f;

static constexpr int B_   = 8;
static constexpr int N_   = 2048;
static constexpr int DIN_ = 512;
static constexpr int DOUT_= 256;
static constexpr int NBKT = 20;

#define WMMA_BF16(a,b,c) \
  __builtin_amdgcn_wmma_f32_16x16x32_bf16(false,(a),false,(b),(short)0,(c),false,false)

__device__ __forceinline__ v16bf pack16(v8bf lo, v8bf hi){
  v16bf r;
  #pragma unroll
  for(int i=0;i<8;++i){ r[i]=lo[i]; r[i+8]=hi[i]; }
  return r;
}

// --------------------------- kernel 0: W -> bf16 ----------------------------
__global__ void convert_f32_bf16(const float* __restrict__ src,
                                 bf16* __restrict__ dst, int n){
  int i = blockIdx.x * blockDim.x + threadIdx.x;
  if(i < n) dst[i] = (bf16)src[i];
}

// ----------------------- kernel 1: fused QKV projection ---------------------
// grid: (B*N/16), block: 128 (4 waves). Wave w computes rows blockIdx.x*16..+15
// x cols w*64..+63 of q, k and v simultaneously (A fragment shared by 12 WMMAs).
__global__ __launch_bounds__(128) void qkv_proj(
    const float* __restrict__ x,
    const bf16* __restrict__ wqb, const bf16* __restrict__ wkb,
    const bf16* __restrict__ wvb,
    const float* __restrict__ bq, const float* __restrict__ bk,
    const float* __restrict__ bv,
    bf16* __restrict__ qbuf, bf16* __restrict__ kbuf, bf16* __restrict__ vtbuf)
{
  __shared__ __align__(16) bf16 vtr[4][64 * 16];   // per-wave V^T staging

  const int lane = threadIdx.x & 31;
  const int wid  = threadIdx.x >> 5;
  const int hi   = lane >> 4;
  const int lm   = lane & 15;
  const int row0 = blockIdx.x * 16;    // global token row (b*N + n)
  const int col0 = wid * 64;

  const v8f zero8 = {0.f,0.f,0.f,0.f,0.f,0.f,0.f,0.f};
  v8f accQ[4] = {zero8, zero8, zero8, zero8};
  v8f accK[4] = {zero8, zero8, zero8, zero8};
  v8f accV[4] = {zero8, zero8, zero8, zero8};

  const float* xp    = x + (size_t)(row0 + lm) * DIN_;
  const int    koffA = hi * 16;
  const size_t wrow  = (size_t)(col0 + lm) * DIN_;  // +t*16*DIN_ per tile

  #pragma unroll
  for(int kb2 = 0; kb2 < DIN_; kb2 += 32){
    // A fragment: x tile 16x32, fp32 -> bf16, contiguous k = 16*hi + s (same as B)
    v16bf a;
    {
      const float* p0 = xp + kb2 + koffA;
      const float* p1 = xp + kb2 + 8 + koffA;
      #pragma unroll
      for(int i=0;i<8;++i){ a[i] = (bf16)p0[i]; a[i+8] = (bf16)p1[i]; }
    }
    #pragma unroll
    for(int t=0;t<4;++t){
      const size_t off = wrow + (size_t)t*16*DIN_ + kb2 + hi*16;
      v16bf bqf = pack16(*(const v8bf*)(wqb + off), *(const v8bf*)(wqb + off + 8));
      v16bf bkf = pack16(*(const v8bf*)(wkb + off), *(const v8bf*)(wkb + off + 8));
      v16bf bvf = pack16(*(const v8bf*)(wvb + off), *(const v8bf*)(wvb + off + 8));
      accQ[t] = WMMA_BF16(a, bqf, accQ[t]);
      accK[t] = WMMA_BF16(a, bkf, accK[t]);
      accV[t] = WMMA_BF16(a, bvf, accV[t]);
    }
  }

  const int bidx = row0 / N_;
  const int nloc = row0 - bidx * N_;
  const float qscale = 0.0625f;        // 1/sqrt(256) folded into q

  // ---- Q / K: row-major stores (lanes = consecutive cols, coalesced b16) ----
  #pragma unroll
  for(int t=0;t<4;++t){
    const int col = col0 + t*16 + lm;
    const float biq = bq[col], bik = bk[col], biv = bv[col];
    #pragma unroll
    for(int r=0;r<8;++r){
      const int m = r + 8*hi;          // C layout: row = r + 8*(lane>=16)
      qbuf[(size_t)(row0 + m)*DOUT_ + col] = (bf16)((accQ[t][r] + biq) * qscale);
      kbuf[(size_t)(row0 + m)*DOUT_ + col] = (bf16)(accK[t][r] + bik);
      // stage V transposed into LDS: vtr[col_local][m]
      vtr[wid][(t*16 + lm)*16 + m] = (bf16)(accV[t][r] + biv);
    }
  }

  // ---- V^T: coalesced b128 stores from LDS staging ----
  asm volatile("s_wait_dscnt 0x0" ::: "memory");
  {
    bf16* vbase = vtbuf + (size_t)bidx*DOUT_*N_ + nloc;
    #pragma unroll
    for(int s=0;s<2;++s){
      const int tr = lane + s*32;                    // transposed row = col_local
      const bf16* lsrc = &vtr[wid][tr*16];
      bf16* gdst = vbase + (size_t)(col0 + tr)*N_;
      *(v8bf*)(gdst)     = *(const v8bf*)(lsrc);     // m = 0..7
      *(v8bf*)(gdst + 8) = *(const v8bf*)(lsrc + 8); // m = 8..15
    }
  }
}

// --------------------------- kernel 2: attention ----------------------------
// grid: (N/64, B), block 128 (4 waves), each wave owns one 16-query tile.
__global__ __launch_bounds__(128) void flash_attn(
    const bf16* __restrict__ qbuf, const bf16* __restrict__ kbuf,
    const bf16* __restrict__ vtbuf,
    const int*  __restrict__ price_rank,   // int32 (JAX default demotes i64)
    const float* __restrict__ Ws, const float* __restrict__ bs,
    const float* __restrict__ rank_emb, const float* __restrict__ rank_w,
    float* __restrict__ out)
{
  __shared__ float gate_lut[NBKT];
  __shared__ __align__(16) bf16 pbuf[4 * 16 * 32];   // per-wave P tile [16][32]

  if(threadIdx.x < NBKT){
    const float z = rank_w[0] * rank_emb[threadIdx.x];
    gate_lut[threadIdx.x] = 1.f / (1.f + __expf(-z));
  }
  __syncthreads();

  const int lane = threadIdx.x & 31;
  const int wid  = threadIdx.x >> 5;
  const int hi   = lane >> 4;
  const int lm   = lane & 15;
  const int bidx = blockIdx.y;
  const int q0   = (blockIdx.x * 4 + wid) * 16;      // query row within batch

  const bf16* qp    = qbuf  + ((size_t)bidx*N_ + q0 + lm) * DOUT_;  // q row lm
  const bf16* krows = kbuf  + (size_t)bidx * N_ * DOUT_;
  const bf16* vt    = vtbuf + (size_t)bidx * DOUT_ * N_;
  const int*  pr    = price_rank + bidx * N_;

  auto loadA = [&](int kk) -> v16bf {
    const bf16* p0 = qp + kk*32 + hi*16;
    return pack16(*(const v8bf*)p0, *(const v8bf*)(p0 + 8));
  };
  auto loadK = [&](int n0, int kk, int sub) -> v16bf {
    const bf16* kp = krows + (size_t)(n0 + sub*16 + lm)*DOUT_ + kk*32 + hi*16;
    return pack16(*(const v8bf*)kp, *(const v8bf*)(kp + 8));
  };
  auto loadV = [&](int n0, int t) -> v16bf {
    const bf16* vp = vt + (size_t)(t*16 + lm)*N_ + n0 + hi*16;
    return pack16(*(const v8bf*)vp, *(const v8bf*)(vp + 8));
  };

  int prq[8];
  #pragma unroll
  for(int r=0;r<8;++r) prq[r] = pr[q0 + r + 8*hi];

  const v8f zero8 = {0.f,0.f,0.f,0.f,0.f,0.f,0.f,0.f};
  v8f ctx[16];
  #pragma unroll
  for(int t=0;t<16;++t) ctx[t] = zero8;
  float mrun[8], lrun[8];
  #pragma unroll
  for(int r=0;r<8;++r){ mrun[r] = -3.0e38f; lrun[r] = 0.f; }

  bf16* myp = pbuf + wid * (16*32);

  for(int n0 = 0; n0 < N_; n0 += 32){
    // prefetch next key block into cache (global_prefetch_b8)
    if(n0 + 32 < N_){
      __builtin_prefetch(krows + (size_t)(n0 + 32 + lm)*DOUT_, 0, 0);
      __builtin_prefetch(krows + (size_t)(n0 + 48 + lm)*DOUT_, 0, 0);
    }

    // ---- S = q @ k^T for 16x32 keys, double-buffered fragments ----
    v8f c0 = zero8, c1 = zero8;
    v16bf aC  = loadA(0);
    v16bf b0C = loadK(n0, 0, 0);
    v16bf b1C = loadK(n0, 0, 1);
    #pragma unroll
    for(int kk=0;kk<8;++kk){
      v16bf aN = aC, b0N = b0C, b1N = b1C;
      if(kk < 7){
        aN  = loadA(kk+1);
        b0N = loadK(n0, kk+1, 0);
        b1N = loadK(n0, kk+1, 1);
      }
      c0 = WMMA_BF16(aC, b0C, c0);
      c1 = WMMA_BF16(aC, b1C, c1);
      aC = aN; b0C = b0N; b1C = b1N;
    }

    // ---- gate + online softmax (per row m = r + 8*hi) ----
    const int prk0 = pr[n0 + lm];
    const int prk1 = pr[n0 + 16 + lm];
    float alpha[8];
    #pragma unroll
    for(int r=0;r<8;++r){
      int d0 = prq[r] - prk0; d0 = d0 < 0 ? -d0 : d0;
      int d1 = prq[r] - prk1; d1 = d1 < 0 ? -d1 : d1;
      int k0i = d0 / 5; if(k0i > NBKT-1) k0i = NBKT-1;
      int k1i = d1 / 5; if(k1i > NBKT-1) k1i = NBKT-1;
      const float s0 = c0[r] * gate_lut[k0i];
      const float s1 = c1[r] * gate_lut[k1i];

      float mx = fmaxf(s0, s1);
      #pragma unroll
      for(int off=1; off<16; off<<=1) mx = fmaxf(mx, __shfl_xor(mx, off, 32));
      const float mnew = fmaxf(mrun[r], mx);
      const float p0 = __expf(s0 - mnew);
      const float p1 = __expf(s1 - mnew);
      const float al = __expf(mrun[r] - mnew);
      float rs = p0 + p1;
      #pragma unroll
      for(int off=1; off<16; off<<=1) rs += __shfl_xor(rs, off, 32);
      lrun[r] = lrun[r] * al + rs;
      mrun[r] = mnew;
      alpha[r] = al;

      bf16* pl = myp + (r + 8*hi)*32 + lm;   // row-major P [16][32]
      pl[0]  = (bf16)p0;
      pl[16] = (bf16)p1;
    }

    // rescale running context
    #pragma unroll
    for(int t=0;t<16;++t){
      #pragma unroll
      for(int r=0;r<8;++r) ctx[t][r] *= alpha[r];
    }

    // make our own LDS stores visible to the ds_load_b128 below
    asm volatile("s_wait_dscnt 0x0" ::: "memory");

    // ---- P A-fragment from LDS, ctx += P @ V (double-buffered V frags) ----
    const bf16* pbA = myp + lm*32 + hi*16;
    v16bf ap = pack16(*(const v8bf*)pbA, *(const v8bf*)(pbA + 8));

    v16bf bvC = loadV(n0, 0);
    #pragma unroll
    for(int t=0;t<16;++t){
      v16bf bvN = bvC;
      if(t < 15) bvN = loadV(n0, t+1);
      ctx[t] = WMMA_BF16(ap, bvC, ctx[t]);
      bvC = bvN;
    }
  }

  // ---- head: winner = sigmoid( (ctx/l) . Ws + bs ) ----
  float wsr[16];
  #pragma unroll
  for(int t=0;t<16;++t) wsr[t] = Ws[t*16 + lm];
  const float bs0 = bs[0];

  #pragma unroll
  for(int r=0;r<8;++r){
    float acc = 0.f;
    #pragma unroll
    for(int t=0;t<16;++t) acc += ctx[t][r] * wsr[t];
    acc /= lrun[r];
    #pragma unroll
    for(int off=1; off<16; off<<=1) acc += __shfl_xor(acc, off, 32);
    const float w = 1.f / (1.f + __expf(-(acc + bs0)));
    if(lm == 0) out[bidx*N_ + q0 + r + 8*hi] = w;
  }
}

// ------------------------------- launcher -----------------------------------
extern "C" void kernel_launch(void* const* d_in, const int* in_sizes, int n_in,
                              void* d_out, int out_size, void* d_ws, size_t ws_size,
                              hipStream_t stream){
  const float* x   = (const float*)d_in[0];
  const int*   prk = (const int*)  d_in[1];
  const float* Wq  = (const float*)d_in[2];
  const float* bq  = (const float*)d_in[3];
  const float* Wk  = (const float*)d_in[4];
  const float* bk  = (const float*)d_in[5];
  const float* Wv  = (const float*)d_in[6];
  const float* bv  = (const float*)d_in[7];
  const float* Ws  = (const float*)d_in[8];
  const float* bs  = (const float*)d_in[9];
  const float* remb= (const float*)d_in[10];
  const float* rw  = (const float*)d_in[11];
  float* out = (float*)d_out;

  const size_t WN = (size_t)DOUT_ * DIN_;     // 131072
  const size_t QN = (size_t)B_ * N_ * DOUT_;  // 4194304
  bf16* wqb = (bf16*)d_ws;
  bf16* wkb = wqb + WN;
  bf16* wvb = wkb + WN;
  bf16* qb  = wvb + WN;
  bf16* kb  = qb  + QN;
  bf16* vtb = kb  + QN;

  const int cthreads = 256;
  convert_f32_bf16<<<((int)WN + cthreads - 1)/cthreads, cthreads, 0, stream>>>(Wq, wqb, (int)WN);
  convert_f32_bf16<<<((int)WN + cthreads - 1)/cthreads, cthreads, 0, stream>>>(Wk, wkb, (int)WN);
  convert_f32_bf16<<<((int)WN + cthreads - 1)/cthreads, cthreads, 0, stream>>>(Wv, wvb, (int)WN);

  qkv_proj<<<dim3((B_*N_)/16), 128, 0, stream>>>(
      x, wqb, wkb, wvb, bq, bk, bv, qb, kb, vtb);

  flash_attn<<<dim3(N_/64, B_), 128, 0, stream>>>(
      qb, kb, vtb, prk, Ws, bs, remb, rw, out);
}
